// VoxelBackBone8xOcc_8134668058640
// MI455X (gfx1250) — matrix operand order repaired and checked by reading, not compiled, not run on hardware
//
#include <hip/hip_runtime.h>

// ---------------------------------------------------------------------------
// CDNA5 (gfx1250) types for WMMA
// ---------------------------------------------------------------------------
typedef __attribute__((ext_vector_type(16))) __bf16 v16bf;
typedef __attribute__((ext_vector_type(8)))  float  v8f;

#define DEV static __device__ __forceinline__

DEV unsigned short f2bf(float f) {
    unsigned int u = __float_as_uint(f);
    unsigned int r = (u + 0x7FFFu + ((u >> 16) & 1u)) >> 16;   // RNE
    return (unsigned short)r;
}

union BF16Vec {                 // 32 bytes: one WMMA bf16 operand per lane
    uint4          q[2];
    unsigned int   u32[8];
    unsigned short u16[16];
    v16bf          v;
};

// ---------------------------------------------------------------------------
// Kernel 1: pack f32 NCDHW input -> bf16 channels-last + occupancy mask
// ---------------------------------------------------------------------------
__global__ void pack_input_kernel(const float* __restrict__ x,
                                  unsigned short* __restrict__ act,
                                  unsigned char* __restrict__ mask, int Nvox) {
    int v = blockIdx.x * blockDim.x + threadIdx.x;
    if (v >= Nvox) return;
    float s = 0.f;
#pragma unroll
    for (int c = 0; c < 4; c++) {
        float f = x[(size_t)c * Nvox + v];
        act[(size_t)v * 4 + c] = f2bf(f);
        s += fabsf(f);
    }
    mask[v] = (s > 0.f) ? 1 : 0;
}

// ---------------------------------------------------------------------------
// Kernel 2: repack weights f32 [Co][Ci][KD][KH][KW] into WMMA A-matrix tiles.
// K order: k = ((kz*KH+ky)*KW+kx)*Ci + ci, zero-padded to NKTA*32 (NKTA has
// two extra zero tiles so the pipelined conv loop can load unconditionally).
// Packed layout: uint index = ((mtile*NKTA + kt)*32 + lane)*8 + r (2 bf16/uint)
// in contiguous k order: element s of lane half side is k = 16*side + s, as in B.
// ---------------------------------------------------------------------------
struct PackWP {
    const float* w;
    unsigned int* wpk;
    int Co, Ci, KD, KH, KW, KHW, K, NKTA;
    long total;
};
__global__ void pack_weights_kernel(PackWP p) {
    long i = (long)blockIdx.x * blockDim.x + threadIdx.x;
    if (i >= p.total) return;
    int  r     = (int)(i & 7);
    int  lane  = (int)((i >> 3) & 31);
    long grp   = i >> 8;                    // mtile*NKTA + kt
    int  kt    = (int)(grp % p.NKTA);
    int  mtile = (int)(grp / p.NKTA);
    int  mloc  = lane & 15, side = lane >> 4;
    int  co    = mtile * 16 + mloc;
    unsigned int out = 0;
#pragma unroll
    for (int h = 0; h < 2; h++) {
        int kk = 2 * r + h + 16 * side;
        int k  = kt * 32 + kk;
        float f = 0.f;
        if (k < p.K && co < p.Co) {
            int kpos = k / p.Ci, ci = k % p.Ci;
            int kz = kpos / p.KHW; int r2 = kpos - kz * p.KHW;
            int ky = r2 / p.KW;    int kx = r2 - ky * p.KW;
            f = p.w[((((size_t)co * p.Ci + ci) * p.KD + kz) * p.KH + ky) * p.KW + kx];
        }
        out |= ((unsigned int)f2bf(f)) << (16 * h);
    }
    p.wpk[i] = out;
}

// ---------------------------------------------------------------------------
// Kernel 3a: implicit-GEMM conv3d, Ci in {16,32,64}, software pipelined.
// One wave/block; wave owns M-tile (16 out ch) x 4 N-tiles (64 voxels).
// Straight-line ping-pong loop (NKTe even, zero A pad). OOB gathers select a
// 64B zero-guard pointer instead of branching, so every B operand is defined
// directly by its global_load_b128 pair (no phis, no register copies).
// ---------------------------------------------------------------------------
struct ConvP {
    const unsigned short* act;     // bf16 channels-last input
    const unsigned int*   wpk;     // packed A tiles
    const unsigned short* zeroblk; // 64B of zeros (OOB guard)
    float*                y;       // f32 channels-last output
    int Ci, ciShift, Co, K, NKTe, NKTA;
    int Di, Hi, Wi, Ho, Wo;
    int KH, KW;
    int sd, sh, sw, pd, ph, pw;
    int Nvox;
};

DEV void load_a(BF16Vec& a, const unsigned int* ap) {
    a.q[0] = ((const uint4*)ap)[0];
    a.q[1] = ((const uint4*)ap)[1];
}

DEV void gather4(const ConvP& p, int kbase, int kz, int ky, int kx, int cb,
                 const int zo[4], const int yo[4], const int xo[4],
                 const bool valid[4], BF16Vec b[4]) {
    bool kok = kbase < p.K;
#pragma unroll
    for (int t = 0; t < 4; t++) {
        int zi = zo[t] * p.sd - p.pd + kz;
        int yi = yo[t] * p.sh - p.ph + ky;
        int xi = xo[t] * p.sw - p.pw + kx;
        bool ok = kok && valid[t] &&
                  (unsigned)zi < (unsigned)p.Di &&
                  (unsigned)yi < (unsigned)p.Hi &&
                  (unsigned)xi < (unsigned)p.Wi;
        const unsigned short* src = ok
            ? (p.act + (((size_t)((zi * p.Hi + yi) * p.Wi + xi)) << p.ciShift) + cb)
            : p.zeroblk;
        b[t].q[0] = ((const uint4*)src)[0];      // unconditional: no phi
        b[t].q[1] = ((const uint4*)src)[1];
    }
}

DEV void wmma4(v8f acc[4], const BF16Vec& a, const BF16Vec b[4]) {
#pragma unroll
    for (int t = 0; t < 4; t++)
        acc[t] = __builtin_amdgcn_wmma_f32_16x16x32_bf16(
            false, a.v, false, b[t].v, (short)0, acc[t], false, false);
}

__global__ __launch_bounds__(32) void conv3d_wmma_fast(ConvP p) {
    const int lane  = threadIdx.x;
    const int n     = lane & 15, side = lane >> 4;
    const int mtile = blockIdx.y;
    const int base  = blockIdx.x * 64;

    v8f acc[4];
#pragma unroll
    for (int t = 0; t < 4; t++)
#pragma unroll
        for (int r = 0; r < 8; r++) acc[t][r] = 0.f;

    int zo[4], yo[4], xo[4];
    bool valid[4];
#pragma unroll
    for (int t = 0; t < 4; t++) {
        int v = base + t * 16 + n;
        valid[t] = v < p.Nvox;
        int vv = valid[t] ? v : 0;
        xo[t] = vv % p.Wo; int tt = vv / p.Wo;
        yo[t] = tt % p.Ho; zo[t] = tt / p.Ho;
    }

    // Incremental K-decode state for this lane's 16-wide slice.
    int kbase = side * 16;
    int cb    = kbase & (p.Ci - 1);
    int kz = 0, ky = 0, kx = 0;
    if ((kbase >> p.ciShift) != 0) {            // initial kpos is 0 or 1
        int kx2 = kx + 1;
        int wx  = (kx2 == p.KW) ? 1 : 0;
        kx = wx ? 0 : kx2;
        int ky2 = ky + wx;
        int wy  = (ky2 == p.KH) ? 1 : 0;
        ky = wy ? 0 : ky2;
        kz += wy;
    }

    const unsigned int* ap =
        p.wpk + ((size_t)mtile * p.NKTA * 32 + (size_t)lane) * 8;

    auto advance = [&]() {                      // kbase += 32, branchless decode
        kbase += 32;
        int c2 = cb + 32;
        int steps = c2 >> p.ciShift;            // 0, 1 or 2
        cb = c2 & (p.Ci - 1);
#pragma unroll
        for (int s = 0; s < 2; s++) {
            int inc = (s < steps) ? 1 : 0;
            int kx2 = kx + inc;
            int wx  = (kx2 == p.KW) ? 1 : 0;
            kx = wx ? 0 : kx2;
            int ky2 = ky + wx;
            int wy  = (ky2 == p.KH) ? 1 : 0;
            ky = wy ? 0 : ky2;
            kz += wy;
        }
    };

    BF16Vec a0, a1, b0[4], b1[4];
    load_a(a0, ap);
    gather4(p, kbase, kz, ky, kx, cb, zo, yo, xo, valid, b0);

    // NKTe is even; wpk has 2 extra zero tiles, and OOB gathers hit the zero
    // guard, so the body is completely branch-free.
    for (int kt = 0; kt < p.NKTe; kt += 2) {
        advance();
        load_a(a1, ap + (size_t)(kt + 1) * 256);
        __builtin_prefetch(ap + (size_t)(kt + 3) * 256, 0, 1);
        gather4(p, kbase, kz, ky, kx, cb, zo, yo, xo, valid, b1);
        wmma4(acc, a0, b0);

        advance();
        load_a(a0, ap + (size_t)(kt + 2) * 256);
        gather4(p, kbase, kz, ky, kx, cb, zo, yo, xo, valid, b0);
        wmma4(acc, a1, b1);
    }

    // D layout: VGPR r -> M = r + 8*side, N = lane%16 => 8 consecutive channels
#pragma unroll
    for (int t = 0; t < 4; t++) {
        int v = base + t * 16 + n;
        if (v < p.Nvox) {
            float* yp = p.y + (size_t)v * p.Co + mtile * 16 + side * 8;
            ((float4*)yp)[0] = make_float4(acc[t][0], acc[t][1], acc[t][2], acc[t][3]);
            ((float4*)yp)[1] = make_float4(acc[t][4], acc[t][5], acc[t][6], acc[t][7]);
        }
    }
}

// ---------------------------------------------------------------------------
// Kernel 3b: first layer (Ci=4, Co=16, 3x3x3, stride 1, pad 1, K=108)
// All decode divisors are compile-time constants; OOB -> zero-guard pointer.
// ---------------------------------------------------------------------------
__global__ __launch_bounds__(32) void conv3d_wmma_c4(
    const unsigned short* __restrict__ act, const unsigned int* __restrict__ wpk,
    const unsigned short* __restrict__ zeroblk,
    float* __restrict__ y, int Di, int Hi, int Wi, int Nvox) {
    const int lane = threadIdx.x;
    const int n = lane & 15, side = lane >> 4;
    const int base = blockIdx.x * 64;

    v8f acc[4];
#pragma unroll
    for (int t = 0; t < 4; t++)
#pragma unroll
        for (int r = 0; r < 8; r++) acc[t][r] = 0.f;

    int zo[4], yo[4], xo[4];
    bool valid[4];
#pragma unroll
    for (int t = 0; t < 4; t++) {
        int v = base + t * 16 + n;
        valid[t] = v < Nvox;
        int vv = valid[t] ? v : 0;
        xo[t] = vv % Wi; int tt = vv / Wi;
        yo[t] = tt % Hi; zo[t] = tt / Hi;
    }

    const unsigned int* ap = wpk + (size_t)lane * 8;
    for (int kt = 0; kt < 4; kt++) {
        BF16Vec a;
        a.q[0] = ((const uint4*)(ap + kt * 256))[0];
        a.q[1] = ((const uint4*)(ap + kt * 256))[1];
        BF16Vec b[4];
#pragma unroll
        for (int t = 0; t < 4; t++) {
#pragma unroll
            for (int j = 0; j < 16; j++) {
                int k = kt * 32 + side * 16 + j;
                int kpos = k >> 2, ci = k & 3;
                int kz = kpos / 9, r2 = kpos - kz * 9;
                int ky = r2 / 3,  kx = r2 - ky * 3;
                int zi = zo[t] - 1 + kz;
                int yi = yo[t] - 1 + ky;
                int xi = xo[t] - 1 + kx;
                bool ok = valid[t] && k < 108 &&
                          (unsigned)zi < (unsigned)Di &&
                          (unsigned)yi < (unsigned)Hi &&
                          (unsigned)xi < (unsigned)Wi;
                const unsigned short* sp = ok
                    ? (act + (((size_t)((zi * Hi + yi) * Wi + xi)) << 2) + ci)
                    : zeroblk;
                b[t].u16[j] = *sp;
            }
        }
#pragma unroll
        for (int t = 0; t < 4; t++)
            acc[t] = __builtin_amdgcn_wmma_f32_16x16x32_bf16(
                false, a.v, false, b[t].v, (short)0, acc[t], false, false);
    }

#pragma unroll
    for (int t = 0; t < 4; t++) {
        int v = base + t * 16 + n;
        if (v < Nvox) {
            float* yp = y + (size_t)v * 16 + side * 8;
            ((float4*)yp)[0] = make_float4(acc[t][0], acc[t][1], acc[t][2], acc[t][3]);
            ((float4*)yp)[1] = make_float4(acc[t][4], acc[t][5], acc[t][6], acc[t][7]);
        }
    }
}

// ---------------------------------------------------------------------------
// Kernel 4: mask downsample  m2 = (conv(m, ones, s, p) > 0)
// ---------------------------------------------------------------------------
__global__ void mask_down_kernel(const unsigned char* __restrict__ mi,
                                 unsigned char* __restrict__ mo,
                                 int Di, int Hi, int Wi, int Do, int Ho, int Wo,
                                 int KD, int KH, int KW, int sd, int sh, int sw,
                                 int pd, int ph, int pw) {
    int v = blockIdx.x * blockDim.x + threadIdx.x;
    int N = Do * Ho * Wo;
    if (v >= N) return;
    int xo = v % Wo; int t = v / Wo;
    int yo = t % Ho; int zo = t / Ho;
    unsigned char r = 0;
    for (int kz = 0; kz < KD && !r; kz++) {
        int zi = zo * sd - pd + kz;
        if ((unsigned)zi >= (unsigned)Di) continue;
        for (int ky = 0; ky < KH && !r; ky++) {
            int yi = yo * sh - ph + ky;
            if ((unsigned)yi >= (unsigned)Hi) continue;
            for (int kx = 0; kx < KW; kx++) {
                int xi = xo * sw - pw + kx;
                if ((unsigned)xi >= (unsigned)Wi) continue;
                if (mi[((size_t)zi * Hi + yi) * Wi + xi]) { r = 1; break; }
            }
        }
    }
    mo[v] = r;
}

// ---------------------------------------------------------------------------
// Kernel 5: masked BN statistics. stats: [0..C)=sum, [C..2C)=sumsq, [2C]=count
// ---------------------------------------------------------------------------
__global__ void bn_stats_kernel(const float* __restrict__ y,
                                const unsigned char* __restrict__ m,
                                int Nvox, int C, float* __restrict__ stats) {
    __shared__ float ssum[128];
    __shared__ float ssq[128];
    __shared__ float scnt;
    for (int c = threadIdx.x; c < C; c += blockDim.x) { ssum[c] = 0.f; ssq[c] = 0.f; }
    if (threadIdx.x == 0) scnt = 0.f;
    __syncthreads();
    float lc = 0.f;
    for (long v = (long)blockIdx.x * blockDim.x + threadIdx.x; v < Nvox;
         v += (long)gridDim.x * blockDim.x) {
        if (m[v]) {
            lc += 1.f;
            const float* yp = y + (size_t)v * C;
            for (int c = 0; c < C; c++) {
                float val = yp[c];
                atomicAdd(&ssum[c], val);
                atomicAdd(&ssq[c], val * val);
            }
        }
    }
    atomicAdd(&scnt, lc);
    __syncthreads();
    for (int c = threadIdx.x; c < C; c += blockDim.x) {
        atomicAdd(&stats[c], ssum[c]);
        atomicAdd(&stats[C + c], ssq[c]);
    }
    if (threadIdx.x == 0) atomicAdd(&stats[2 * C], scnt);
}

// ---------------------------------------------------------------------------
// Kernel 6: BN normalize + ReLU + mask; emit bf16 act (next layer) or f32 NCDHW
// ---------------------------------------------------------------------------
__global__ void bn_apply_kernel(const float* __restrict__ y,
                                const unsigned char* __restrict__ m,
                                const float* __restrict__ stats,
                                const float* __restrict__ g,
                                const float* __restrict__ b,
                                int Nvox, int C,
                                unsigned short* __restrict__ actout,
                                float* __restrict__ fout) {
    long i = (long)blockIdx.x * blockDim.x + threadIdx.x;
    long total = (long)Nvox * C;
    if (i >= total) return;
    int  c = (int)(i % C);
    long v = i / C;
    float cnt = stats[2 * C];
    cnt = cnt < 1.f ? 1.f : cnt;
    float mean = stats[c] / cnt;
    float var  = stats[C + c] / cnt - mean * mean;
    float val  = (y[i] - mean) * rsqrtf(var + 1e-3f) * g[c] + b[c];
    val = fmaxf(val, 0.f);
    val = m[v] ? val : 0.f;
    if (actout) actout[i] = f2bf(val);
    if (fout)   fout[(size_t)c * Nvox + v] = val;
}

// ---------------------------------------------------------------------------
// Host orchestration
// ---------------------------------------------------------------------------
struct LayerCfg { int Ci, Co, KD, KH, KW, sd, sh, sw, pd, ph, pw, subm; };

extern "C" void kernel_launch(void* const* d_in, const int* in_sizes, int n_in,
                              void* d_out, int out_size, void* d_ws, size_t ws_size,
                              hipStream_t stream) {
    (void)in_sizes; (void)n_in; (void)out_size;
    static const LayerCfg L[12] = {
        {  4,  16, 3, 3, 3, 1, 1, 1, 1, 1, 1, 1 },
        { 16,  16, 3, 3, 3, 1, 1, 1, 1, 1, 1, 1 },
        { 16,  32, 3, 3, 3, 2, 2, 2, 1, 1, 1, 0 },
        { 32,  32, 3, 3, 3, 1, 1, 1, 1, 1, 1, 1 },
        { 32,  32, 3, 3, 3, 1, 1, 1, 1, 1, 1, 1 },
        { 32,  64, 3, 3, 3, 2, 2, 2, 1, 1, 1, 0 },
        { 64,  64, 3, 3, 3, 1, 1, 1, 1, 1, 1, 1 },
        { 64,  64, 3, 3, 3, 1, 1, 1, 1, 1, 1, 1 },
        { 64,  64, 3, 3, 3, 2, 2, 2, 0, 1, 1, 0 },
        { 64,  64, 3, 3, 3, 1, 1, 1, 1, 1, 1, 1 },
        { 64,  64, 3, 3, 3, 1, 1, 1, 1, 1, 1, 1 },
        { 64, 128, 3, 1, 1, 2, 1, 1, 0, 0, 0, 0 },
    };

    const size_t MAXVOX = 41UL * 352UL * 400UL;          // 5,772,800
    char*  ws  = (char*)d_ws;
    size_t off = 0;
    auto carve = [&](size_t bytes) -> void* {
        void* pp = ws + off;
        off = (off + bytes + 255) & ~(size_t)255;
        return pp;
    };

    unsigned short* actA  = (unsigned short*)carve(MAXVOX * 16 * 2);
    unsigned short* actB  = (unsigned short*)carve(MAXVOX * 16 * 2);
    float*          ybuf  = (float*)carve(MAXVOX * 16 * 4);
    unsigned char*  mbase = (unsigned char*)carve(6700000);
    float*          stats = (float*)carve(1536);
    unsigned short* zeroblk = (unsigned short*)carve(256);

    unsigned int* wpk[12];
    int Kl[12], NKTe[12], NKTA[12];
    for (int i = 0; i < 12; i++) {
        Kl[i]   = L[i].KD * L[i].KH * L[i].KW * L[i].Ci;
        int nkt = (Kl[i] + 31) / 32;
        NKTe[i] = (nkt + 1) & ~1;                // even trip count
        NKTA[i] = NKTe[i] + 2;                   // + zero pad tiles
        wpk[i]  = (unsigned int*)carve((size_t)(L[i].Co / 16) * NKTA[i] * 32 * 8 * 4);
    }
    if (off > ws_size) return;   // insufficient workspace: bail safely

    const float* x = (const float*)d_in[0];

    hipMemsetAsync(zeroblk, 0, 256, stream);     // OOB zero guard
    pack_input_kernel<<<(int)((MAXVOX + 255) / 256), 256, 0, stream>>>(
        x, actA, mbase, (int)MAXVOX);
    for (int i = 0; i < 12; i++) {
        PackWP pw;
        pw.w = (const float*)d_in[1 + 3 * i];
        pw.wpk = wpk[i];
        pw.Co = L[i].Co; pw.Ci = L[i].Ci;
        pw.KD = L[i].KD; pw.KH = L[i].KH; pw.KW = L[i].KW;
        pw.KHW = L[i].KH * L[i].KW;
        pw.K = Kl[i]; pw.NKTA = NKTA[i];
        pw.total = (long)(L[i].Co / 16) * NKTA[i] * 256;
        pack_weights_kernel<<<(int)((pw.total + 255) / 256), 256, 0, stream>>>(pw);
    }

    unsigned short* cur = actA;
    unsigned short* nxt = actB;
    unsigned char*  curMask = mbase;
    size_t maskOff = MAXVOX;
    int D = 41, H = 352, W = 400;

    for (int i = 0; i < 12; i++) {
        const LayerCfg c = L[i];
        int Do = D, Ho = H, Wo = W;
        unsigned char* outMask = curMask;
        if (!c.subm) {
            Do = (D + 2 * c.pd - c.KD) / c.sd + 1;
            Ho = (H + 2 * c.ph - c.KH) / c.sh + 1;
            Wo = (W + 2 * c.pw - c.KW) / c.sw + 1;
            outMask = mbase + maskOff;
            maskOff += (size_t)Do * Ho * Wo;
            int N = Do * Ho * Wo;
            mask_down_kernel<<<(N + 255) / 256, 256, 0, stream>>>(
                curMask, outMask, D, H, W, Do, Ho, Wo,
                c.KD, c.KH, c.KW, c.sd, c.sh, c.sw, c.pd, c.ph, c.pw);
        }
        int Nvox = Do * Ho * Wo;

        if (i == 0) {
            conv3d_wmma_c4<<<dim3((Nvox + 63) / 64, 1, 1), 32, 0, stream>>>(
                cur, wpk[0], zeroblk, ybuf, D, H, W, Nvox);
        } else {
            ConvP cp;
            cp.act = cur; cp.wpk = wpk[i]; cp.zeroblk = zeroblk; cp.y = ybuf;
            cp.Ci = c.Ci;
            cp.ciShift = 31 - __builtin_clz((unsigned)c.Ci);
            cp.Co = c.Co; cp.K = Kl[i]; cp.NKTe = NKTe[i]; cp.NKTA = NKTA[i];
            cp.Di = D; cp.Hi = H; cp.Wi = W;
            cp.Ho = Ho; cp.Wo = Wo;
            cp.KH = c.KH; cp.KW = c.KW;
            cp.sd = c.sd; cp.sh = c.sh; cp.sw = c.sw;
            cp.pd = c.pd; cp.ph = c.ph; cp.pw = c.pw;
            cp.Nvox = Nvox;
            conv3d_wmma_fast<<<dim3((Nvox + 63) / 64, c.Co / 16, 1), 32, 0, stream>>>(cp);
        }

        hipMemsetAsync(stats, 0, (size_t)(2 * c.Co + 1) * sizeof(float), stream);
        bn_stats_kernel<<<1024, 256, 0, stream>>>(ybuf, outMask, Nvox, c.Co, stats);

        const float* g = (const float*)d_in[2 + 3 * i];
        const float* b = (const float*)d_in[3 + 3 * i];
        long total = (long)Nvox * c.Co;
        bool last = (i == 11);
        bn_apply_kernel<<<(int)((total + 255) / 256), 256, 0, stream>>>(
            ybuf, outMask, stats, g, b, Nvox, c.Co,
            last ? nullptr : nxt, last ? (float*)d_out : nullptr);

        unsigned short* tmp = cur; cur = nxt; nxt = tmp;
        D = Do; H = Ho; W = Wo;
        curMask = outMask;
    }
}
